// VectorizedConstellationAttention_55671366091366
// MI455X (gfx1250) — compile-verified, matrix operand order checked by reading, not run on hardware
//
#include <hip/hip_runtime.h>
#include <hip/hip_bf16.h>
#include <math.h>

// Problem constants (from reference)
#define BB   2
#define TT   2048
#define DDIM 1024
#define HALF 512
#define KSEL 15
#define HIDN 64
#define MTOT (BB * TT)          // 4096 rows total

typedef __attribute__((ext_vector_type(16))) __bf16 v16bf;
typedef __attribute__((ext_vector_type(8)))  float  v8f;

union FragBF { v16bf v; unsigned int u[8]; };

// Per ISA 7.12.2: 16-bit A-matrix 16x32. Lane l holds row m=l&15.
// kb = (l>=16)?8:0 ; VGPR v holds K pair starting at:
__device__ __forceinline__ int frag_k0(int v, int kb) {
  return ((v & 4) ? 16 : 0) + kb + (v & 3) * 2;
}

// ---------------------------------------------------------------- staging paths
#if __has_builtin(__builtin_amdgcn_global_load_async_to_lds_b128)
#define USE_ASYNC_LDS 1
#else
#define USE_ASYNC_LDS 0
#endif

#if __has_builtin(__builtin_amdgcn_tensor_load_to_lds)
#define USE_TDM 1
#else
#define USE_TDM 0
#endif

typedef int v4i_vs __attribute__((vector_size(4 * sizeof(int))));
typedef __attribute__((address_space(1))) v4i_vs as1_v4i;
typedef __attribute__((address_space(3))) v4i_vs as3_v4i;
typedef __attribute__((address_space(3))) void  as3_void;

__device__ __forceinline__ void cp16_to_lds(const __bf16* g, __bf16* l) {
#if USE_ASYNC_LDS
  __builtin_amdgcn_global_load_async_to_lds_b128(
      (as1_v4i*)(void*)const_cast<__bf16*>(g),
      (as3_v4i*)(void*)l, 0, 0);
#else
  *(int4*)l = *(const int4*)g;    // global_load_b128 + ds_store_b128
#endif
}

#if USE_TDM
typedef unsigned int u32x4 __attribute__((vector_size(16)));
typedef int          i32x4 __attribute__((vector_size(16)));
typedef int          i32x8 __attribute__((vector_size(32)));

__device__ __forceinline__ unsigned lds_byte_off(void* p) {
  return (unsigned)(unsigned long long)(as3_void*)p;
}

// TDM D# for a 2D bf16 tile: tile_dim0=32 (K), tile_dim1=rows, row stride 1024
// elements, written contiguously (row*32 elems) at lds_addr. ISA ch.8 layout.
__device__ __forceinline__ void tdm_load_tile(unsigned long long gaddr,
                                              unsigned lds_addr,
                                              unsigned rows,
                                              unsigned tensor_rows) {
  u32x4 g0 = {};
  g0[0] = 1u;                                               // count=1 (valid D#)
  g0[1] = lds_addr;                                         // LDS byte address
  g0[2] = (unsigned)(gaddr & 0xFFFFFFFFu);                  // global_addr[31:0]
  g0[3] = (unsigned)((gaddr >> 32) & 0x01FFFFFFu) | (2u << 30);  // [56:32] | type=2
  i32x8 g1 = {};
  g1[0] = (int)(1u << 16);                                  // data_size=1 (2 bytes)
  g1[1] = (int)((1024u & 0xFFFFu) << 16);                   // tensor_dim0 = 1024
  g1[2] = (int)(((tensor_rows & 0xFFFFu) << 16) | ((1024u >> 16) & 0xFFFFu));
  g1[3] = (int)(((32u & 0xFFFFu) << 16) | ((tensor_rows >> 16) & 0xFFFFu)); // tile_dim0=32
  g1[4] = (int)(rows & 0xFFFFu);                            // tile_dim1 = rows
  g1[5] = 1024;                                             // tensor_dim0_stride lo
  g1[6] = 0;
  g1[7] = 0;
  i32x4 z4 = {};
#if __clang_major__ >= 23
  i32x8 z8 = {};
  __builtin_amdgcn_tensor_load_to_lds(g0, g1, z4, z4, z8, 0);
#else
  __builtin_amdgcn_tensor_load_to_lds(g0, g1, z4, z4, 0);
#endif
}
#endif

__device__ __forceinline__ void stage_wait() {
#if USE_TDM
#if __has_builtin(__builtin_amdgcn_s_wait_tensorcnt)
  __builtin_amdgcn_s_wait_tensorcnt(0);
#else
  asm volatile("s_wait_tensorcnt 0x0" ::: "memory");
#endif
#endif
#if USE_ASYNC_LDS
#if __has_builtin(__builtin_amdgcn_s_wait_asynccnt)
  __builtin_amdgcn_s_wait_asynccnt(0);
#else
  asm volatile("s_wait_asynccnt 0x0" ::: "memory");
#endif
#endif
  __syncthreads();
}

// ---------------------------------------------------------------- casts
__global__ void cast_bf16_kernel(const float* __restrict__ in,
                                 __bf16* __restrict__ out, int n) {
  int i = blockIdx.x * 256 + threadIdx.x;
  if (i < n) out[i] = (__bf16)in[i];
}

// out[n*D + k] = (bf16) in[k*D + n]   (N-major weight for B-fragment loads)
__global__ void transpose_bf16_kernel(const float* __restrict__ in,
                                      __bf16* __restrict__ out) {
  int i = blockIdx.x * 256 + threadIdx.x;
  if (i < DDIM * DDIM) {
    int n = i / DDIM, k = i % DDIM;
    out[i] = (__bf16)in[k * DDIM + n];
  }
}

// ---------------------------------------------------------------- LDS-staged GEMM
// C(MxN f32) = A(MxK bf16, row-major, MTOT rows) @ B(KxN), B given as BT (N-major).
// Block = 256 threads (8 waves): 64M x 128N x 32K tile, double-buffered LDS.
// A tile staged by the Tensor Data Mover (one tensor_load_to_lds per k-step,
// issued by wave 0, TENSORcnt); B tile staged per-lane with
// global_load_async_to_lds_b128 (ASYNCcnt). Waves 4(M) x 2(N), 16x64 each.
__global__ void __launch_bounds__(256)
gemm_lds_kernel(const __bf16* __restrict__ A, const __bf16* __restrict__ BT,
                float* __restrict__ C) {
  __shared__ __bf16 Al[2][64 * 32];
  __shared__ __bf16 Bl[2][128 * 32];
  const int tid = threadIdx.x;
  const int l   = tid & 31;
  const int w   = tid >> 5;
  const int mw  = w & 3, nw = w >> 2;
  const int m0  = blockIdx.x * 64;
  const int n0  = blockIdx.y * 128;

  // per-thread 16B staging chunks for the B tile: 128x32 bf16 = 512 x 16B
  const int b0r = (tid * 2) >> 2,     b0c = ((tid * 2) & 3) * 8;
  const int b1r = (tid * 2 + 1) >> 2, b1c = ((tid * 2 + 1) & 3) * 8;
#if !USE_TDM
  const int ar  = tid >> 2, ac = (tid & 3) * 8;   // A tile fallback staging
#endif

  v8f acc[4] = {};
  const int kb   = (l >> 4) * 8;
  const int mloc = mw * 16 + (l & 15);

  // ---- prologue: fill buffer 0
#if USE_TDM
  if (w == 0)
    tdm_load_tile((unsigned long long)(const void*)A + ((size_t)m0 * DDIM + 0) * 2,
                  lds_byte_off(&Al[0][0]), 64u, (unsigned)MTOT);
#else
  cp16_to_lds(A + (size_t)(m0 + ar) * DDIM + 0 + ac, &Al[0][ar * 32 + ac]);
#endif
  cp16_to_lds(BT + (size_t)(n0 + b0r) * DDIM + 0 + b0c, &Bl[0][b0r * 32 + b0c]);
  cp16_to_lds(BT + (size_t)(n0 + b1r) * DDIM + 0 + b1c, &Bl[0][b1r * 32 + b1c]);
  stage_wait();

  int buf = 0;
  for (int ks = 0; ks < DDIM; ks += 32) {
    if (ks + 32 < DDIM) {
      const int kn = ks + 32, nb = buf ^ 1;
#if USE_TDM
      if (w == 0)
        tdm_load_tile((unsigned long long)(const void*)A + ((size_t)m0 * DDIM + kn) * 2,
                      lds_byte_off(&Al[nb][0]), 64u, (unsigned)MTOT);
#else
      cp16_to_lds(A + (size_t)(m0 + ar) * DDIM + kn + ac, &Al[nb][ar * 32 + ac]);
#endif
      cp16_to_lds(BT + (size_t)(n0 + b0r) * DDIM + kn + b0c, &Bl[nb][b0r * 32 + b0c]);
      cp16_to_lds(BT + (size_t)(n0 + b1r) * DDIM + kn + b1c, &Bl[nb][b1r * 32 + b1c]);
    }
    // load A-frag and all four B-frags, then issue WMMAs back-to-back (XDL stream)
    FragBF a;
#pragma unroll
    for (int v = 0; v < 8; ++v)
      a.u[v] = *(const unsigned int*)(&Al[buf][mloc * 32 + frag_k0(v, kb)]);
    FragBF fb[4];
#pragma unroll
    for (int ns = 0; ns < 4; ++ns) {
      const int nloc = nw * 64 + ns * 16 + (l & 15);
#pragma unroll
      for (int v = 0; v < 8; ++v)
        fb[ns].u[v] = *(const unsigned int*)(&Bl[buf][nloc * 32 + frag_k0(v, kb)]);
    }
#pragma unroll
    for (int ns = 0; ns < 4; ++ns)
      acc[ns] = __builtin_amdgcn_wmma_f32_16x16x32_bf16(false, a.v, false, fb[ns].v,
                                                        (short)0, acc[ns], false, false);
    stage_wait();   // next buffer landed (TENSORcnt+ASYNCcnt) AND all waves done reading
    buf ^= 1;
  }
#pragma unroll
  for (int ns = 0; ns < 4; ++ns)
#pragma unroll
    for (int r = 0; r < 8; ++r) {
      int m = m0 + mw * 16 + r + ((l >> 4) * 8);
      int n = n0 + nw * 64 + ns * 16 + (l & 15);
      C[(size_t)m * DDIM + n] = acc[ns][r];
    }
}

// ---------------------------------------------------------------- RoPE + cast
__global__ void rope_cast_kernel(float* __restrict__ Ipre, float* __restrict__ Ppre,
                                 __bf16* __restrict__ Ib, __bf16* __restrict__ Pb) {
  int i = blockIdx.x * 256 + threadIdx.x;      // over B*T*HALF
  if (i >= BB * TT * HALF) return;
  int d = i % HALF;
  int row = i / HALF;          // b*T + t
  int t = row % TT;
  float inv = __expf(-(float)d * (9.210340371976184f / (float)HALF));
  float ang = (float)t * inv;
  float s = __sinf(ang), c = __cosf(ang);
  size_t base = (size_t)row * DDIM;
  float i1 = Ipre[base + d], i2 = Ipre[base + d + HALF];
  float o1 = i1 * c - i2 * s, o2 = i1 * s + i2 * c;
  Ipre[base + d] = o1; Ipre[base + d + HALF] = o2;
  Ib[base + d] = (__bf16)o1; Ib[base + d + HALF] = (__bf16)o2;
  float p1 = Ppre[base + d], p2 = Ppre[base + d + HALF];
  float q1 = p1 * c - p2 * s, q2 = p1 * s + p2 * c;
  Ppre[base + d] = q1; Ppre[base + d + HALF] = q2;
  Pb[base + d] = (__bf16)q1; Pb[base + d + HALF] = (__bf16)q2;
}

// ---------------------------------------------------------------- logits + top-k
// grid = (TT/16, BB), block = 32. One wave handles 16 query rows; streams
// 16x16 score tiles (K=1024 -> 32 WMMA each), keeps per-row top-15 in regs.
// (Top-k is kept unsorted, in slot order; the features MLP input rel[] follows that order.)
__global__ void logits_topk_kernel(const __bf16* __restrict__ Ib,
                                   const __bf16* __restrict__ Pb,
                                   float* __restrict__ tkv, int* __restrict__ tki) {
  __shared__ float tile[16][17];
  const int b  = blockIdx.y;
  const int t0 = blockIdx.x * 16;
  const int l  = threadIdx.x;
  const __bf16* Ibb = Ib + (size_t)b * TT * DDIM;
  const __bf16* Pbb = Pb + (size_t)b * TT * DDIM;
  const int mrow = t0 + (l & 15);
  const int kb   = (l >> 4) * 8;

  float best[KSEL];
  int   bidx[KSEL];
#pragma unroll
  for (int k = 0; k < KSEL; ++k) { best[k] = -__builtin_inff(); bidx[k] = 0; }

  for (int s0 = 0; s0 <= t0; s0 += 16) {
    v8f acc = {};
    for (int ks = 0; ks < DDIM; ks += 32) {
      FragBF a, fb;
#pragma unroll
      for (int v = 0; v < 8; ++v) {
        int k0 = frag_k0(v, kb);
        a.u[v]  = *(const unsigned int*)(Ibb + (size_t)mrow * DDIM + ks + k0);
        fb.u[v] = *(const unsigned int*)(Pbb + (size_t)(s0 + (l & 15)) * DDIM + ks + k0);
      }
      acc = __builtin_amdgcn_wmma_f32_16x16x32_bf16(false, a.v, false, fb.v,
                                                    (short)0, acc, false, false);
    }
#pragma unroll
    for (int r = 0; r < 8; ++r)
      tile[r + ((l >> 4) * 8)][l & 15] = acc[r];
    __syncthreads();
    if (l < 16) {
      const int t = t0 + l;
#pragma unroll
      for (int j = 0; j < 16; ++j) {
        int s = s0 + j;
        if (s <= t) {
          float v = tile[l][j] * 0.03125f;   // D^-0.5
          float mn = best[0]; int mp = 0;
#pragma unroll
          for (int q = 1; q < KSEL; ++q) if (best[q] < mn) { mn = best[q]; mp = q; }
          if (v > mn) {
#pragma unroll
            for (int q = 0; q < KSEL; ++q) if (q == mp) { best[q] = v; bidx[q] = s; }
          }
        }
      }
    }
    __syncthreads();
  }
  if (l < 16) {
    const int t = t0 + l;
    size_t base = ((size_t)b * TT + t) * KSEL;
#pragma unroll
    for (int k = 0; k < KSEL; ++k) { tkv[base + k] = best[k]; tki[base + k] = bidx[k]; }
  }
}

__device__ __forceinline__ float gelu_exact(float x) {
  return 0.5f * x * (1.0f + erff(x * 0.70710678118654752f));
}

// ---------------------------------------------------------------- features
// grid = B*T, block = 32. One wave per (b,t).
// Gram of Q = [P_sel rows 0..14 ; I row] via WMMA with A-frag == B-frag
// (dot-then-normalize == normalize-then-dot). Then 15 lanes run the MLP.
__global__ void features_kernel(const __bf16* __restrict__ Ib,
                                const __bf16* __restrict__ Pb,
                                const int* __restrict__ tki,
                                const float* __restrict__ tkv,
                                const float* __restrict__ W1, const float* __restrict__ b1,
                                const float* __restrict__ W2, const float* __restrict__ b2,
                                const float* __restrict__ Wc, const float* __restrict__ bc,
                                const float* __restrict__ Wm, const float* __restrict__ bm,
                                const float* __restrict__ palette,
                                __bf16* __restrict__ Vb) {
  const int bt = blockIdx.x;
  const int b  = bt >> 11;
  const int t  = bt & (TT - 1);
  const int l  = threadIdx.x;

  __shared__ int   sel[16];
  __shared__ float keep[16];
  __shared__ float gmat[16][17];
  __shared__ float nrm[16];
  __shared__ float mixs[16];
  __shared__ float wq[KSEL][4];
  __shared__ int   oq[KSEL][4];

  if (l < KSEL) {
    sel[l]  = tki[(size_t)bt * KSEL + l];
    keep[l] = (tkv[(size_t)bt * KSEL + l] > -1e30f) ? 1.0f : 0.0f;
  }
  if (l == KSEL) { sel[KSEL] = t; keep[KSEL] = 1.0f; mixs[KSEL] = -__builtin_inff(); }
  __syncthreads();

  // Gram via WMMA: lane l reads row (l&15) of Q.
  const int m  = l & 15;
  const int kb = (l >> 4) * 8;
  const __bf16* rowp = (m == KSEL)
      ? (Ib + ((size_t)b * TT + t) * DDIM)
      : (Pb + ((size_t)b * TT + sel[m]) * DDIM);
  v8f acc = {};
  for (int ks = 0; ks < DDIM; ks += 32) {
    FragBF a;
#pragma unroll
    for (int v = 0; v < 8; ++v)
      a.u[v] = *(const unsigned int*)(rowp + ks + frag_k0(v, kb));
    acc = __builtin_amdgcn_wmma_f32_16x16x32_bf16(false, a.v, false, a.v,
                                                  (short)0, acc, false, false);
  }
#pragma unroll
  for (int r = 0; r < 8; ++r)
    gmat[r + ((l >> 4) * 8)][l & 15] = acc[r];
  __syncthreads();
  if (l < 16) nrm[l] = fmaxf(sqrtf(fmaxf(gmat[l][l], 0.0f)), 1e-12f);
  __syncthreads();

  float zx = 0.0f, zy = 0.0f, mix = -__builtin_inff();
  if (l < KSEL) {
    float rel[KSEL + 2];
    const float nk = nrm[l];
#pragma unroll
    for (int j = 0; j < KSEL; ++j) {
      float gv = gmat[l][j] / (nk * nrm[j]);
      gv = fminf(fmaxf(gv, -1.0f), 1.0f);
      rel[j] = gv * keep[l] * keep[j];
    }
    float fa = gmat[KSEL][l] / (nrm[KSEL] * nk);
    rel[KSEL] = fminf(fmaxf(fa, -1.0f), 1.0f) * keep[l];
    rel[KSEL + 1] = fmaxf((float)(t - sel[l]), 0.0f) * (1.0f / (float)TT) * keep[l];

    float h1[HIDN];
    for (int h = 0; h < HIDN; ++h) {
      float s = b1[h];
      for (int i = 0; i < KSEL + 2; ++i) s += rel[i] * W1[i * HIDN + h];
      h1[h] = gelu_exact(s);
    }
    float h2[HIDN];
    for (int h = 0; h < HIDN; ++h) {
      float s = b2[h];
      for (int i = 0; i < HIDN; ++i) s += h1[i] * W2[i * HIDN + h];
      h2[h] = gelu_exact(s);
    }
    float z0 = bc[0], z1 = bc[1], mm = bm[0];
    for (int i = 0; i < HIDN; ++i) {
      z0 += h2[i] * Wc[i * 2 + 0];
      z1 += h2[i] * Wc[i * 2 + 1];
      mm += h2[i] * Wm[i];
    }
    zx = tanhf(z0); zy = tanhf(z1);
    mix = (keep[l] > 0.0f) ? mm : -__builtin_inff();
    mixs[l] = mix;
  }
  __syncthreads();

  if (l < KSEL) {
    float mx = -__builtin_inff();
    for (int q = 0; q < KSEL; ++q) mx = fmaxf(mx, mixs[q]);
    float ssum = 0.0f;
    for (int q = 0; q < KSEL; ++q)
      ssum += (mixs[q] > -1e30f) ? __expf(mixs[q] - mx) : 0.0f;
    float wk = (mix > -1e30f) ? (__expf(mix - mx) / ssum) : 0.0f;

    float fx = fminf(fmaxf((zx + 1.0f) * 0.5f * 15.0f, 0.0f), 15.0f);
    float fy = fminf(fmaxf((zy + 1.0f) * 0.5f * 15.0f, 0.0f), 15.0f);
    int x0 = (int)floorf(fx); int x1 = min(x0 + 1, 15);
    int y0 = (int)floorf(fy); int y1 = min(y0 + 1, 15);
    float wx = fx - (float)x0, wy = fy - (float)y0;
    oq[l][0] = y0 * 16 + x0; wq[l][0] = wk * (1.0f - wy) * (1.0f - wx);
    oq[l][1] = y0 * 16 + x1; wq[l][1] = wk * (1.0f - wy) * wx;
    oq[l][2] = y1 * 16 + x0; wq[l][2] = wk * wy * (1.0f - wx);
    oq[l][3] = y1 * 16 + x1; wq[l][3] = wk * wy * wx;
  }
  __syncthreads();

  for (int d = l; d < DDIM; d += 32) {
    const float* pd = palette + (size_t)d * 256;
    float a = 0.0f;
#pragma unroll
    for (int k = 0; k < KSEL; ++k) {
      a += wq[k][0] * pd[oq[k][0]];
      a += wq[k][1] * pd[oq[k][1]];
      a += wq[k][2] * pd[oq[k][2]];
      a += wq[k][3] * pd[oq[k][3]];
    }
    Vb[(size_t)bt * DDIM + d] = (__bf16)a;
  }
}

// ---------------------------------------------------------------- launch
extern "C" void kernel_launch(void* const* d_in, const int* in_sizes, int n_in,
                              void* d_out, int out_size, void* d_ws, size_t ws_size,
                              hipStream_t stream) {
  const float* x   = (const float*)d_in[0];
  const float* Wi  = (const float*)d_in[1];
  const float* Wp  = (const float*)d_in[2];
  const float* pal = (const float*)d_in[3];
  const float* W1  = (const float*)d_in[4];
  const float* b1  = (const float*)d_in[5];
  const float* W2  = (const float*)d_in[6];
  const float* b2  = (const float*)d_in[7];
  const float* Wc  = (const float*)d_in[8];
  const float* bc  = (const float*)d_in[9];
  const float* Wm  = (const float*)d_in[10];
  const float* bm  = (const float*)d_in[11];
  const float* Wo  = (const float*)d_in[12];
  float* y = (float*)d_out;

  char* ws = (char*)d_ws;
  size_t off = 0;
  auto take = [&](size_t bytes) {
    void* p = ws + off;
    off = (off + bytes + 255) & ~(size_t)255;
    return p;
  };
  __bf16* xb   = (__bf16*)take((size_t)MTOT * DDIM * 2);
  __bf16* WiT  = (__bf16*)take((size_t)DDIM * DDIM * 2);
  __bf16* WpT  = (__bf16*)take((size_t)DDIM * DDIM * 2);
  __bf16* WoT  = (__bf16*)take((size_t)DDIM * DDIM * 2);
  float*  Ipre = (float*)take((size_t)MTOT * DDIM * 4);
  float*  Ppre = (float*)take((size_t)MTOT * DDIM * 4);
  __bf16* Ibf  = (__bf16*)take((size_t)MTOT * DDIM * 2);
  __bf16* Pbf  = (__bf16*)take((size_t)MTOT * DDIM * 2);
  float*  tkv  = (float*)take((size_t)MTOT * KSEL * 4);
  int*    tki  = (int*)take((size_t)MTOT * KSEL * 4);
  __bf16* Vb   = (__bf16*)take((size_t)MTOT * DDIM * 2);

  // 1) casts + weight transposes (bf16 operands for WMMA)
  cast_bf16_kernel<<<(MTOT * DDIM + 255) / 256, 256, 0, stream>>>(x, xb, MTOT * DDIM);
  transpose_bf16_kernel<<<(DDIM * DDIM + 255) / 256, 256, 0, stream>>>(Wi, WiT);
  transpose_bf16_kernel<<<(DDIM * DDIM + 255) / 256, 256, 0, stream>>>(Wp, WpT);
  transpose_bf16_kernel<<<(DDIM * DDIM + 255) / 256, 256, 0, stream>>>(Wo, WoT);

  // 2) I_pre = x@Wi, P_pre = x@Wp  (TDM + async-LDS staged WMMA GEMMs)
  gemm_lds_kernel<<<dim3(MTOT / 64, DDIM / 128), 256, 0, stream>>>(xb, WiT, Ipre);
  gemm_lds_kernel<<<dim3(MTOT / 64, DDIM / 128), 256, 0, stream>>>(xb, WpT, Ppre);

  // 3) RoPE in place + bf16 copies
  rope_cast_kernel<<<(BB * TT * HALF + 255) / 256, 256, 0, stream>>>(Ipre, Ppre, Ibf, Pbf);

  // 4) causal logits + top-15 (WMMA score tiles)
  logits_topk_kernel<<<dim3(TT / 16, BB), 32, 0, stream>>>(Ibf, Pbf, tkv, tki);

  // 5) Gram(WMMA) + MLP + softmax + bilinear palette mix -> V (bf16)
  features_kernel<<<MTOT, 32, 0, stream>>>(Ibf, Pbf, tki, tkv,
                                           W1, b1, W2, b2, Wc, bc, Wm, bm,
                                           pal, Vb);

  // 6) y = V @ Wo[0]
  gemm_lds_kernel<<<dim3(MTOT / 64, DDIM / 128), 256, 0, stream>>>(Vb, WoT, y);
}
